// DummyTransformer_42545946034479
// MI455X (gfx1250) — 8-bit matrix operand order repaired and checked by reading, compiled, not run on hardware
//
#include <hip/hip_runtime.h>
#include <stdint.h>
#include <stddef.h>


typedef __attribute__((ext_vector_type(8))) int v8i;
typedef __attribute__((ext_vector_type(4))) int vi4;

#define BATCH 4
#define SEQ   1024
#define EMB   1024
#define NH    16
#define HD    64
#define FFN   4096
#define NROW  (BATCH * SEQ)

#ifndef __has_builtin
#define __has_builtin(x) 0
#endif
#if __has_builtin(__builtin_amdgcn_tensor_load_to_lds) && \
    __has_builtin(__builtin_amdgcn_s_wait_tensorcnt)
#define HAVE_TDM 1
#else
#define HAVE_TDM 0
#endif
#if __has_builtin(__builtin_amdgcn_global_load_async_to_lds_b128) && \
    __has_builtin(__builtin_amdgcn_s_wait_asynccnt)
#define HAVE_ASYNC 1
#else
#define HAVE_ASYNC 0
#endif

#define GAS1(p) ((__attribute__((address_space(1))) vi4*)((void*)(p)))
#define LAS3(p) ((__attribute__((address_space(3))) vi4*)((void*)(p)))

static __device__ __forceinline__ v8i zero8() {
  v8i z = {0, 0, 0, 0, 0, 0, 0, 0};
  return z;
}

static __device__ __forceinline__ v8i wmma_iu8(v8i a, v8i b, v8i c) {
  return __builtin_amdgcn_wmma_i32_16x16x64_iu8(true, a, true, b, c, false, false);
}

static __device__ __forceinline__ v8i frag_a(const char* base, int ld, int lane) {
  const int m  = lane & 15;
  const int hi = (lane >> 4) & 1;
  v8i f;
#pragma unroll
  for (int d = 0; d < 8; ++d) {
    const int k = ((d >> 1) << 4) + (hi << 3) + ((d & 1) << 2);
    f[d] = *(const int*)(base + m * ld + k);
  }
  return f;
}

static __device__ __forceinline__ v8i frag_b(const char* base, int ld, int lane) {
  const int n  = lane & 15;
  const int hi = (lane >> 4) & 1;
  v8i f;
#pragma unroll
  for (int v = 0; v < 8; ++v) {
    const int k = ((v >> 1) << 4) + (hi << 3) + ((v & 1) << 2);
    f[v] = *(const int*)(base + n * ld + k);
  }
  return f;
}

#if HAVE_TDM
typedef __attribute__((ext_vector_type(4))) unsigned tdm_u4;
typedef __attribute__((ext_vector_type(4))) int      tdm_i4;
typedef __attribute__((ext_vector_type(8))) int      tdm_i8;

static __device__ __forceinline__ unsigned lds_off(void* p) {
  return (unsigned)(unsigned long long)(__attribute__((address_space(3))) char*)p;
}

static __device__ __forceinline__ void tdm_stage(const void* gtile, unsigned ldsOff,
                                                 unsigned K, unsigned rowsTot) {
  const unsigned long long ga = (unsigned long long)(uintptr_t)gtile;
  tdm_u4 g0;
  g0[0] = 1u;
  g0[1] = ldsOff;
  g0[2] = (unsigned)ga;
  g0[3] = (unsigned)((ga >> 32) & 0x01ffffffull) | (2u << 30);
  tdm_i8 g1;
  g1[0] = 0;
  g1[1] = (int)((K & 0xffffu) << 16);
  g1[2] = (int)((K >> 16) | ((rowsTot & 0xffffu) << 16));
  g1[3] = (int)(((rowsTot >> 16) & 0xffffu) | (64u << 16));
  g1[4] = 128;
  g1[5] = (int)K;
  g1[6] = 0;
  g1[7] = 0;
  tdm_i4 z4 = {0, 0, 0, 0};
#if __clang_major__ >= 23
  tdm_i8 z8 = {0, 0, 0, 0, 0, 0, 0, 0};
  __builtin_amdgcn_tensor_load_to_lds(g0, g1, z4, z4, z8, 0);
#else
  __builtin_amdgcn_tensor_load_to_lds(g0, g1, z4, z4, 0);
#endif
}
#endif

static __device__ __forceinline__ void gemm_stage_wait() {
#if HAVE_TDM
  __builtin_amdgcn_s_wait_tensorcnt(0);
#elif HAVE_ASYNC
  __builtin_amdgcn_s_wait_asynccnt(0);
#endif
}

__global__ void __launch_bounds__(256)
k_absmax(const float* __restrict__ x, size_t n, unsigned* amax_bits, int posOnly) {
  float m = 0.0f;
  for (size_t i = (size_t)blockIdx.x * blockDim.x + threadIdx.x; i < n;
       i += (size_t)gridDim.x * blockDim.x) {
    const float v = x[i];
    m = fmaxf(m, posOnly ? v : fabsf(v));
  }
#pragma unroll
  for (int off = 16; off > 0; off >>= 1) m = fmaxf(m, __shfl_xor(m, off, 32));
  __shared__ float sm[8];
  if ((threadIdx.x & 31) == 0) sm[threadIdx.x >> 5] = m;
  __syncthreads();
  if (threadIdx.x == 0) {
    float r = sm[0];
#pragma unroll
    for (int w = 1; w < 8; ++w) r = fmaxf(r, sm[w]);
    atomicMax(amax_bits, __float_as_uint(fmaxf(r, 0.0f)));
  }
}

__global__ void __launch_bounds__(256)
k_quant(const float* __restrict__ x, int8_t* __restrict__ q,
        const unsigned* amax_bits, float premul, float* scale_out, size_t n) {
  const float amax = __uint_as_float(*amax_bits) * premul;
  const float s = fmaxf(amax * (1.0f / 127.0f), 1e-8f);
  if (blockIdx.x == 0 && threadIdx.x == 0) *scale_out = s;
  const float inv = premul / s;
  for (size_t i = (size_t)blockIdx.x * blockDim.x + threadIdx.x; i < n;
       i += (size_t)gridDim.x * blockDim.x) {
    int qi = __float2int_rn(x[i] * inv);
    qi = qi > 127 ? 127 : (qi < -127 ? -127 : qi);
    q[i] = (int8_t)qi;
  }
}

__global__ void __launch_bounds__(256)
k_fakequant(float* __restrict__ x, const unsigned* amax_bits, int unsignedRelu, size_t n) {
  const float amax = __uint_as_float(*amax_bits);
  const float nl = unsignedRelu ? 255.0f : 127.0f;
  const float s = fmaxf(amax / nl, 1e-8f);
  const float inv = 1.0f / s;
  const float lo = unsignedRelu ? 0.0f : -127.0f;
  for (size_t i = (size_t)blockIdx.x * blockDim.x + threadIdx.x; i < n;
       i += (size_t)gridDim.x * blockDim.x) {
    float v = x[i];
    if (unsignedRelu) v = fmaxf(v, 0.0f);
    float r = rintf(v * inv);
    r = fminf(fmaxf(r, lo), nl);
    x[i] = r * s;
  }
}

__global__ void __launch_bounds__(256)
k_gemm_i8(const int8_t* __restrict__ A, const int8_t* __restrict__ W,
          const float* __restrict__ bias, float* __restrict__ out,
          int Mr, int Nc, int K, const float* sA, const float* sW) {
  __shared__ __attribute__((aligned(16))) char As[2][128 * 64];
  __shared__ __attribute__((aligned(16))) char Bs[2][128 * 64];

  const int tid = threadIdx.x;
  const int lane = tid & 31;
  const int wave = tid >> 5;
  const int hi = (lane >> 4) & 1, ln = lane & 15;
  const int wm = wave >> 1;
  const int wn = wave & 1;
  const int blockM = blockIdx.y * 128;
  const int blockN = blockIdx.x * 128;
  const int srow = tid >> 1;
  const int scol = (tid & 1) * 32;
  (void)Mr; (void)srow; (void)scol;

#if HAVE_TDM
  const unsigned aoff[2] = {lds_off(As[0]), lds_off(As[1])};
  const unsigned boff[2] = {lds_off(Bs[0]), lds_off(Bs[1])};
#endif

  auto stage = [&](int buf, int k0) {
#if HAVE_TDM
    if (wave == 0) {
      tdm_stage(A + (size_t)blockM * K + k0, aoff[buf], (unsigned)K, (unsigned)Mr);
      tdm_stage(W + (size_t)blockN * K + k0, boff[buf], (unsigned)K, (unsigned)Nc);
    }
#elif HAVE_ASYNC
    const int8_t* ga = A + (size_t)(blockM + srow) * K + k0 + scol;
    const int8_t* gb = W + (size_t)(blockN + srow) * K + k0 + scol;
    char* la = As[buf] + srow * 64 + scol;
    char* lb = Bs[buf] + srow * 64 + scol;
    __builtin_amdgcn_global_load_async_to_lds_b128(GAS1(ga), LAS3(la), 0, 0);
    __builtin_amdgcn_global_load_async_to_lds_b128(GAS1(ga + 16), LAS3(la + 16), 0, 0);
    __builtin_amdgcn_global_load_async_to_lds_b128(GAS1(gb), LAS3(lb), 0, 0);
    __builtin_amdgcn_global_load_async_to_lds_b128(GAS1(gb + 16), LAS3(lb + 16), 0, 0);
#else
    const int8_t* ga = A + (size_t)(blockM + srow) * K + k0 + scol;
    const int8_t* gb = W + (size_t)(blockN + srow) * K + k0 + scol;
    char* la = As[buf] + srow * 64 + scol;
    char* lb = Bs[buf] + srow * 64 + scol;
    *(int4*)la = *(const int4*)ga;
    *(int4*)(la + 16) = *(const int4*)(ga + 16);
    *(int4*)lb = *(const int4*)gb;
    *(int4*)(lb + 16) = *(const int4*)(gb + 16);
#endif
  };

  v8i acc[2][4];
#pragma unroll
  for (int i = 0; i < 2; ++i)
#pragma unroll
    for (int j = 0; j < 4; ++j) acc[i][j] = zero8();

  stage(0, 0);
  gemm_stage_wait();
  __syncthreads();

  int buf = 0;
  for (int k0 = 0; k0 < K; k0 += 64) {
    if (k0 + 64 < K) stage(buf ^ 1, k0 + 64);

    const char* ab = As[buf];
    const char* bb = Bs[buf];
    const v8i a0 = frag_a(ab + (wm * 2 + 0) * 16 * 64, 64, lane);
    const v8i a1 = frag_a(ab + (wm * 2 + 1) * 16 * 64, 64, lane);
#pragma unroll
    for (int j = 0; j < 4; ++j) {
      const v8i bj = frag_b(bb + (wn * 4 + j) * 16 * 64, 64, lane);
      acc[0][j] = wmma_iu8(a0, bj, acc[0][j]);
      acc[1][j] = wmma_iu8(a1, bj, acc[1][j]);
    }
    gemm_stage_wait();
    __syncthreads();
    buf ^= 1;
  }

  const float s = sA[0] * sW[0];
  const float invs = 1.0f / s;
#pragma unroll
  for (int j = 0; j < 4; ++j) {
    const int n = blockN + (wn * 4 + j) * 16 + ln;
    float rb = rintf(bias[n] * invs);
    rb = fminf(fmaxf(rb, -127.0f), 127.0f);
    const float qb = rb * s;
#pragma unroll
    for (int i = 0; i < 2; ++i)
#pragma unroll
      for (int r = 0; r < 8; ++r) {
        const int m = blockM + (wm * 2 + i) * 16 + r + hi * 8;
        out[(size_t)m * Nc + n] = (float)acc[i][j][r] * s + qb;
      }
  }
}

__global__ void __launch_bounds__(256)
k_attn(const int8_t* __restrict__ qq, const int8_t* __restrict__ kq,
       const int8_t* __restrict__ vq, float* __restrict__ of,
       const float* s_qs, const float* s_k, const float* s_v) {
  __shared__ __attribute__((aligned(16))) char ks[64 * 64];
  __shared__ __attribute__((aligned(16))) char vT[64 * 64];
  __shared__ __attribute__((aligned(16))) char pbuf[8][16 * 64];

  const int lane = threadIdx.x & 31;
  const int wave = threadIdx.x >> 5;
  const int hi = (lane >> 4) & 1;
  const int ln = lane & 15;
  const int rowBlk = blockIdx.x;
  const int h = blockIdx.y;
  const int b = blockIdx.z;
  const int rowBase = rowBlk * 128 + wave * 16;
  const size_t bh = ((size_t)b * SEQ) * EMB + (size_t)h * HD;
  const int8_t* qg = qq + bh;
  const int8_t* kg = kq + bh;
  const int8_t* vg = vq + bh;

  v8i qfr;
  {
    const char* qrow = (const char*)(qg + (size_t)(rowBase + ln) * EMB);
#pragma unroll
    for (int d = 0; d < 8; ++d) {
      const int k = ((d >> 1) << 4) + (hi << 3) + ((d & 1) << 2);
      qfr[d] = *(const int*)(qrow + k);
    }
  }

  const float ss = s_qs[0] * s_k[0];
  const int nJG = (rowBlk + 1) * 2;

  float mrow[8], lrow[8];
#pragma unroll
  for (int r = 0; r < 8; ++r) { mrow[r] = -3.0e38f; lrow[r] = 0.0f; }

  const int tkey = threadIdx.x >> 2;
  const int tcol = (threadIdx.x & 3) * 16;

  for (int jg = 0; jg < nJG; ++jg) {
    const int8_t* ksrc = kg + (size_t)(jg * 64 + tkey) * EMB + tcol;
#if HAVE_ASYNC
    __builtin_amdgcn_global_load_async_to_lds_b128(GAS1(ksrc),
                                                   LAS3(ks + tkey * 64 + tcol), 0, 0);
    __builtin_amdgcn_s_wait_asynccnt(0);
#else
    *(int4*)(ks + tkey * 64 + tcol) = *(const int4*)ksrc;
#endif
    __syncthreads();
#pragma unroll
    for (int js = 0; js < 4; ++js) {
      const v8i kfr = frag_b(ks + js * 16 * 64, 64, lane);
      v8i acc = zero8();
      acc = wmma_iu8(qfr, kfr, acc);
      const int col = jg * 64 + js * 16 + ln;
#pragma unroll
      for (int r = 0; r < 8; ++r) {
        const int row = rowBase + r + hi * 8;
        const float sc = (col <= row) ? ((float)acc[r] * ss) : -3.0e38f;
        const float mo = mrow[r];
        const float mn = fmaxf(mo, sc);
        lrow[r] = lrow[r] * __expf(mo - mn) + __expf(sc - mn);
        mrow[r] = mn;
      }
    }
    __syncthreads();
  }
#pragma unroll
  for (int r = 0; r < 8; ++r) {
    float m = mrow[r], l = lrow[r];
#pragma unroll
    for (int off = 1; off < 16; off <<= 1) {
      const float mo = __shfl_xor(m, off, 32);
      const float lo = __shfl_xor(l, off, 32);
      const float mn = fmaxf(m, mo);
      l = l * __expf(m - mn) + lo * __expf(mo - mn);
      m = mn;
    }
    mrow[r] = m;
    lrow[r] = (l > 0.0f) ? (1.0f / l) : 0.0f;
  }

  v8i oacc[4];
#pragma unroll
  for (int dt = 0; dt < 4; ++dt) oacc[dt] = zero8();

  for (int jg = 0; jg < nJG; ++jg) {
    const int8_t* ksrc = kg + (size_t)(jg * 64 + tkey) * EMB + tcol;
#if HAVE_ASYNC
    __builtin_amdgcn_global_load_async_to_lds_b128(GAS1(ksrc),
                                                   LAS3(ks + tkey * 64 + tcol), 0, 0);
#else
    *(int4*)(ks + tkey * 64 + tcol) = *(const int4*)ksrc;
#endif
    {
      const int4 vv = *(const int4*)(vg + (size_t)(jg * 64 + tkey) * EMB + tcol);
      const char* vb = (const char*)&vv;
#pragma unroll
      for (int e = 0; e < 16; ++e) vT[(tcol + e) * 64 + tkey] = vb[e];
    }
#if HAVE_ASYNC
    __builtin_amdgcn_s_wait_asynccnt(0);
#endif
    __syncthreads();
#pragma unroll
    for (int js = 0; js < 4; ++js) {
      const v8i kfr = frag_b(ks + js * 16 * 64, 64, lane);
      v8i acc = zero8();
      acc = wmma_iu8(qfr, kfr, acc);
      const int col = jg * 64 + js * 16 + ln;
#pragma unroll
      for (int r = 0; r < 8; ++r) {
        const int row = rowBase + r + hi * 8;
        const float sc = (col <= row) ? ((float)acc[r] * ss) : -3.0e38f;
        const float p = __expf(sc - mrow[r]) * lrow[r];
        int pq = __float2int_rn(p * 127.0f);
        pq = pq > 127 ? 127 : (pq < -127 ? -127 : pq);
        pbuf[wave][(r + hi * 8) * 64 + js * 16 + ln] = (char)pq;
      }
    }
    __syncthreads();
    const v8i pfr = frag_a(pbuf[wave], 64, lane);
#pragma unroll
    for (int dt = 0; dt < 4; ++dt) {
      const v8i bfr = frag_b(vT + dt * 16 * 64, 64, lane);
      oacc[dt] = wmma_iu8(pfr, bfr, oacc[dt]);
    }
    __syncthreads();
  }

  const float os = (1.0f / 127.0f) * s_v[0];
#pragma unroll
  for (int dt = 0; dt < 4; ++dt)
#pragma unroll
    for (int r = 0; r < 8; ++r) {
      const int row = rowBase + r + hi * 8;
      of[((size_t)b * SEQ + row) * EMB + h * HD + dt * 16 + ln] =
          (float)oacc[dt][r] * os;
    }
}


enum {
  AW_Q = 0, AW_K, AW_V, AW_O, AW_1, AW_2,
  AX = 8, AQS, AK, AV, AO, AA, AH1, AHPOS, AHQ, AG2, AG2POS
};

extern "C" void kernel_launch(void* const* d_in, const int* in_sizes, int n_in,
                              void* d_out, int out_size, void* d_ws, size_t ws_size,
                              hipStream_t stream) {
  (void)in_sizes; (void)n_in; (void)out_size; (void)ws_size;
  const float* x0 = (const float*)d_in[0];
  const float* Wq = (const float*)d_in[2];
  const float* bq = (const float*)d_in[3];
  const float* Wk = (const float*)d_in[4];
  const float* bk = (const float*)d_in[5];
  const float* Wv = (const float*)d_in[6];
  const float* bv = (const float*)d_in[7];
  const float* Wo = (const float*)d_in[8];
  const float* bo = (const float*)d_in[9];
  const float* W1 = (const float*)d_in[10];
  const float* b1 = (const float*)d_in[11];
  const float* W2 = (const float*)d_in[12];
  const float* b2 = (const float*)d_in[13];

  char* ws = (char*)d_ws;
  size_t off = 0;
  auto take = [&](size_t sz) {
    size_t r = off;
    off += (sz + 255) & ~(size_t)255;
    return ws + r;
  };

  unsigned* amax = (unsigned*)take(64 * sizeof(unsigned));
  float*    scl  = (float*)take(64 * sizeof(float));
  int8_t* wq_q = (int8_t*)take((size_t)EMB * EMB);
  int8_t* wk_q = (int8_t*)take((size_t)EMB * EMB);
  int8_t* wv_q = (int8_t*)take((size_t)EMB * EMB);
  int8_t* wo_q = (int8_t*)take((size_t)EMB * EMB);
  int8_t* w1_q = (int8_t*)take((size_t)FFN * EMB);
  int8_t* w2_q = (int8_t*)take((size_t)EMB * FFN);
  int8_t* xq   = (int8_t*)take((size_t)NROW * EMB);
  float*  qf   = (float*)take((size_t)NROW * EMB * 4);
  float*  kf   = (float*)take((size_t)NROW * EMB * 4);
  float*  vf   = (float*)take((size_t)NROW * EMB * 4);
  int8_t* qq8  = (int8_t*)take((size_t)NROW * EMB);
  int8_t* kq8  = (int8_t*)take((size_t)NROW * EMB);
  int8_t* vq8  = (int8_t*)take((size_t)NROW * EMB);
  float*  of   = (float*)take((size_t)NROW * EMB * 4);
  int8_t* oq8  = (int8_t*)take((size_t)NROW * EMB);
  float*  af   = (float*)take((size_t)NROW * EMB * 4);
  int8_t* aq8  = (int8_t*)take((size_t)NROW * EMB);
  float*  hf   = (float*)take((size_t)NROW * FFN * 4);
  int8_t* hq8  = (int8_t*)take((size_t)NROW * FFN);
  float*  x1   = (float*)take((size_t)NROW * EMB * 4);

  const dim3 blk(256);
  const dim3 rg(1024);

  auto absmax = [&](const float* p, size_t n, int slot, int posOnly) {
    k_absmax<<<rg, blk, 0, stream>>>(p, n, amax + slot, posOnly);
  };
  auto quant = [&](const float* p, int8_t* q, int slot, float premul, size_t n) {
    k_quant<<<rg, blk, 0, stream>>>(p, q, amax + slot, premul, scl + slot, n);
  };
  auto fakeq = [&](float* p, int slot, int uRelu, size_t n) {
    k_fakequant<<<rg, blk, 0, stream>>>(p, amax + slot, uRelu, n);
  };
  auto gemm = [&](const int8_t* A, const int8_t* Wt, const float* bias, float* outp,
                  int Nc, int K, int sa, int sw) {
    k_gemm_i8<<<dim3(Nc / 128, NROW / 128), blk, 0, stream>>>(
        A, Wt, bias, outp, NROW, Nc, K, scl + sa, scl + sw);
  };

  (void)hipMemsetAsync(amax, 0, 64 * sizeof(unsigned), stream);

  const size_t nEE = (size_t)EMB * EMB, nME = (size_t)FFN * EMB;
  absmax(Wq, nEE, AW_Q, 0);  quant(Wq, wq_q, AW_Q, 1.0f, nEE);
  absmax(Wk, nEE, AW_K, 0);  quant(Wk, wk_q, AW_K, 1.0f, nEE);
  absmax(Wv, nEE, AW_V, 0);  quant(Wv, wv_q, AW_V, 1.0f, nEE);
  absmax(Wo, nEE, AW_O, 0);  quant(Wo, wo_q, AW_O, 1.0f, nEE);
  absmax(W1, nME, AW_1, 0);  quant(W1, w1_q, AW_1, 1.0f, nME);
  absmax(W2, nME, AW_2, 0);  quant(W2, w2_q, AW_2, 1.0f, nME);

  const size_t nAct = (size_t)NROW * EMB;
  const size_t nHid = (size_t)NROW * FFN;
  float* lout[2] = { x1, (float*)d_out };
  const float* xin = x0;

  for (int L = 0; L < 2; ++L) {
    (void)hipMemsetAsync(amax + 8, 0, 56 * sizeof(unsigned), stream);

    absmax(xin, nAct, AX, 0);
    quant(xin, xq, AX, 1.0f, nAct);
    gemm(xq, wq_q, bq, qf, EMB, EMB, AX, AW_Q);
    gemm(xq, wk_q, bk, kf, EMB, EMB, AX, AW_K);
    gemm(xq, wv_q, bv, vf, EMB, EMB, AX, AW_V);

    absmax(qf, nAct, AQS, 0);  quant(qf, qq8, AQS, 0.125f, nAct);
    absmax(kf, nAct, AK, 0);   quant(kf, kq8, AK, 1.0f, nAct);
    absmax(vf, nAct, AV, 0);   quant(vf, vq8, AV, 1.0f, nAct);

    k_attn<<<dim3(SEQ / 128, NH, BATCH), blk, 0, stream>>>(
        qq8, kq8, vq8, of, scl + AQS, scl + AK, scl + AV);

    absmax(of, nAct, AO, 0);   quant(of, oq8, AO, 1.0f, nAct);
    gemm(oq8, wo_q, bo, af, EMB, EMB, AO, AW_O);

    absmax(af, nAct, AA, 0);   quant(af, aq8, AA, 1.0f, nAct);
    gemm(aq8, w1_q, b1, hf, FFN, EMB, AA, AW_1);

    absmax(hf, nHid, AH1, 0);    fakeq(hf, AH1, 0, nHid);
    absmax(hf, nHid, AHPOS, 1);  fakeq(hf, AHPOS, 1, nHid);
    absmax(hf, nHid, AHQ, 0);    quant(hf, hq8, AHQ, 1.0f, nHid);
    gemm(hq8, w2_q, b2, lout[L], EMB, FFN, AHQ, AW_2);

    absmax(lout[L], nAct, AG2, 0);    fakeq(lout[L], AG2, 0, nAct);
    absmax(lout[L], nAct, AG2POS, 1); fakeq(lout[L], AG2POS, 1, nAct);
    xin = lout[L];
  }
}
